// TransformerEncoder_25950192403158
// MI455X (gfx1250) — matrix operand order repaired and checked by reading, not compiled, not run on hardware
//
#include <hip/hip_runtime.h>
#include <hip/hip_bf16.h>

#define D     512
#define NHEAD 8
#define DKH   64
#define LSEQ  1024
#define BB    4
#define NROWS (BB * LSEQ)   // 4096
#define DFF   2048
#define NLAYER 6
#define NCLS  10

typedef __attribute__((ext_vector_type(16))) _Float16 v16h;
typedef __attribute__((ext_vector_type(8)))  float    v8f;

// Load a 16-f16 WMMA fragment as two 16-byte chunks (global: global_load_b128,
// LDS: ds_load_b128 after address-space inference).
__device__ __forceinline__ v16h ld_frag(const _Float16* p0, const _Float16* p1) {
  union { uint4 q[2]; v16h v; } u;
  u.q[0] = *(const uint4*)p0;
  u.q[1] = *(const uint4*)p1;
  return u.v;
}

// ---------------------------------------------------------------------------
// Generic WMMA GEMM:  out = ((A @ W^T) + bias) * scale  [relu] [+ res]
// A: (M,K) f16 row-major.  W: (N,K) f16 row-major (so B-frags are contiguous).
// Block = 256 threads (8 waves) in a 4(M) x 2(N) wave grid; block tile
// 128x128, wave tile 32x64 (2 A-frags x 4 B-frags -> 8 WMMAs per K-step).
// Epilogue behavior is compile-time (no per-element branches).
// ---------------------------------------------------------------------------
template <int RELU, int HAS_RES, int OUT_F, int OUT_H>
__global__ __launch_bounds__(256)
void gemm_wmma(const _Float16* __restrict__ A, const _Float16* __restrict__ W,
               const float* __restrict__ bias, const float* __restrict__ res,
               float* __restrict__ outF, _Float16* __restrict__ outH,
               int M, int N, int K, float scale)
{
  const int lane  = threadIdx.x & 31;
  const int wave  = threadIdx.x >> 5;
  const int mBase = blockIdx.y * 128 + (wave & 3) * 32;
  const int nBase = blockIdx.x * 128 + (wave >> 2) * 64;
  const int lrow  = lane & 15;
  const int hihalf = (lane & 16) ? 1 : 0;

  v8f acc[2][4] = {};

  const _Float16* aRow0 = A + (size_t)(mBase + lrow) * K + hihalf * 8;
  const _Float16* aRow1 = aRow0 + (size_t)16 * K;
  const _Float16* wRow  = W + (size_t)(nBase + lrow) * K + hihalf * 8;

  for (int k0 = 0; k0 < K; k0 += 32) {
    v16h a[2], b[4];
    a[0] = ld_frag(aRow0 + k0, aRow0 + k0 + 16);
    a[1] = ld_frag(aRow1 + k0, aRow1 + k0 + 16);
#pragma unroll
    for (int ni = 0; ni < 4; ++ni) {
      const _Float16* wp = wRow + (size_t)(ni * 16) * K + k0;
      __builtin_prefetch(wp + 32, 0, 0);   // global_prefetch_b8 next K-step
      b[ni] = ld_frag(wp, wp + 16);
    }
#pragma unroll
    for (int mi = 0; mi < 2; ++mi)
#pragma unroll
      for (int ni = 0; ni < 4; ++ni)
        acc[mi][ni] = __builtin_amdgcn_wmma_f32_16x16x32_f16(
            false, a[mi], false, b[ni], (short)0, acc[mi][ni], false, false);
  }

#pragma unroll
  for (int mi = 0; mi < 2; ++mi)
#pragma unroll
    for (int ni = 0; ni < 4; ++ni) {
      const int n = nBase + ni * 16 + lrow;
      const float bn = bias[n];
#pragma unroll
      for (int v = 0; v < 8; ++v) {
        const int m = mBase + mi * 16 + v + hihalf * 8;   // C-layout row
        float val = (acc[mi][ni][v] + bn) * scale;
        if (RELU) val = fmaxf(val, 0.0f);
        const size_t idx = (size_t)m * N + n;
        if (HAS_RES) val += res[idx];
        if (OUT_F) outF[idx] = val;
        if (OUT_H) outH[idx] = (_Float16)val;
      }
    }
}

// ---------------------------------------------------------------------------
// Flash attention: one wave per 16-query tile, streams keys 32 at a time.
// Q: (4096,512) f16 (1/sqrt(dk) already folded in). KV: (4096,1024) f16,
// cols [0,512)=K, [512,1024)=V.  VT: (B*H*64, 1024) f16 = V transposed.
// OUT: (4096,512) f16 in (b,l,h,dk) layout.
// ---------------------------------------------------------------------------
__global__ __launch_bounds__(256)
void attn_wmma(const _Float16* __restrict__ Q, const _Float16* __restrict__ KV,
               const _Float16* __restrict__ VT, _Float16* __restrict__ OUT)
{
  __shared__ _Float16 pbuf[8][16 * 40];   // per-wave 16x32 P tile, padded rows

  const int lane = threadIdx.x & 31;
  const int wave = threadIdx.x >> 5;
  const int lrow = lane & 15;
  const int hihalf = (lane & 16) ? 1 : 0;
  const int bh = blockIdx.y;              // 0..31
  const int b  = bh >> 3;
  const int h  = bh & 7;
  const int q0 = blockIdx.x * 128 + wave * 16;   // query tile base in [0,1024)

  // Q fragments (16 rows x 64 d) -> two 16x32 A-frags, kept in registers
  v16h aq[2];
#pragma unroll
  for (int t = 0; t < 2; ++t) {
    const _Float16* qp = Q + (size_t)(b * LSEQ + q0 + lrow) * D + h * DKH + t * 32 + hihalf * 8;
    aq[t] = ld_frag(qp, qp + 16);
  }

  v8f o[4] = {};
  float mrun[8], lrun[8];
#pragma unroll
  for (int v = 0; v < 8; ++v) { mrun[v] = -1e30f; lrun[v] = 0.0f; }

  for (int kc = 0; kc < LSEQ; kc += 32) {
    // S = Q @ K^T for 32 keys: two 16x16 score tiles, each 2 WMMAs over d
    v8f s[2] = {};
#pragma unroll
    for (int j = 0; j < 2; ++j)
#pragma unroll
      for (int t = 0; t < 2; ++t) {
        const _Float16* kp = KV + (size_t)(b * LSEQ + kc + j * 16 + lrow) * (2 * D)
                             + h * DKH + t * 32 + hihalf * 8;
        v16h bk = ld_frag(kp, kp + 16);
        s[j] = __builtin_amdgcn_wmma_f32_16x16x32_f16(
            false, aq[t], false, bk, (short)0, s[j], false, false);
      }

    // Online softmax update. C-layout: row r = v + 8*hihalf spans 16 lanes of
    // one half-wave; xor masks {1,2,4,8} reduce halves independently (wave32).
    float p0[8], p1[8];
#pragma unroll
    for (int v = 0; v < 8; ++v) {
      float mx = fmaxf(s[0][v], s[1][v]);
      mx = fmaxf(mx, __shfl_xor(mx, 1, 32));
      mx = fmaxf(mx, __shfl_xor(mx, 2, 32));
      mx = fmaxf(mx, __shfl_xor(mx, 4, 32));
      mx = fmaxf(mx, __shfl_xor(mx, 8, 32));
      const float mnew = fmaxf(mrun[v], mx);
      const float corr = __expf(mrun[v] - mnew);
      mrun[v] = mnew;
      p0[v] = __expf(s[0][v] - mnew);
      p1[v] = __expf(s[1][v] - mnew);
      float rs = p0[v] + p1[v];
      rs += __shfl_xor(rs, 1, 32);
      rs += __shfl_xor(rs, 2, 32);
      rs += __shfl_xor(rs, 4, 32);
      rs += __shfl_xor(rs, 8, 32);
      lrun[v] = lrun[v] * corr + rs;
#pragma unroll
      for (int ni = 0; ni < 4; ++ni) o[ni][v] *= corr;
    }

    // C-layout -> A-layout transpose of P through per-wave LDS
    _Float16* pw = &pbuf[wave][0];
#pragma unroll
    for (int v = 0; v < 8; ++v) {
      const int r = v + hihalf * 8;
      pw[r * 40 + lrow]      = (_Float16)p0[v];
      pw[r * 40 + 16 + lrow] = (_Float16)p1[v];
    }
    asm volatile("s_wait_dscnt 0" ::: "memory");
    const _Float16* pr = pw + lrow * 40 + hihalf * 8;
    v16h ap = ld_frag(pr, pr + 16);

    // O += P @ V  (V^T rows contiguous along keys -> direct B-frags)
#pragma unroll
    for (int ni = 0; ni < 4; ++ni) {
      const _Float16* vp = VT + (size_t)(bh * DKH + ni * 16 + lrow) * LSEQ + kc + hihalf * 8;
      v16h bv = ld_frag(vp, vp + 16);
      o[ni] = __builtin_amdgcn_wmma_f32_16x16x32_f16(
          false, ap, false, bv, (short)0, o[ni], false, false);
    }
  }

#pragma unroll
  for (int ni = 0; ni < 4; ++ni)
#pragma unroll
    for (int v = 0; v < 8; ++v) {
      const int r = v + hihalf * 8;
      const float val = o[ni][v] / lrun[v];
      OUT[(size_t)(b * LSEQ + q0 + r) * D + h * DKH + ni * 16 + lrow] = (_Float16)val;
    }
}

// ---------------------------------------------------------------------------
// LayerNorm over D=512, one block (256 threads) per row, f16 output.
// ---------------------------------------------------------------------------
__global__ __launch_bounds__(256)
void ln_rows(const float* __restrict__ X, const float* __restrict__ g,
             const float* __restrict__ be, _Float16* __restrict__ outH)
{
  __shared__ float rs[256], rs2[256];
  const int row = blockIdx.x;
  const int t = threadIdx.x;
  const float* xr = X + (size_t)row * D;
  const float a = xr[t];
  const float c = xr[t + 256];
  rs[t]  = a + c;
  rs2[t] = a * a + c * c;
  __syncthreads();
  for (int off = 128; off > 0; off >>= 1) {
    if (t < off) { rs[t] += rs[t + off]; rs2[t] += rs2[t + off]; }
    __syncthreads();
  }
  const float mean = rs[0] * (1.0f / D);
  const float var  = rs2[0] * (1.0f / D) - mean * mean;
  const float inv  = rsqrtf(var + 1e-5f);
  outH[(size_t)row * D + t]       = (_Float16)((a - mean) * inv * g[t]       + be[t]);
  outH[(size_t)row * D + t + 256] = (_Float16)((c - mean) * inv * g[t + 256] + be[t + 256]);
}

// Embedding * sqrt(D) + sinusoidal positional encoding (fp32 master).
__global__ void embed_pe(const int* __restrict__ tok, const float* __restrict__ emb,
                         float* __restrict__ X)
{
  const int idx = blockIdx.x * blockDim.x + threadIdx.x;
  if (idx >= NROWS * D) return;
  const int d   = idx & (D - 1);
  const int row = idx >> 9;
  const int l   = row & (LSEQ - 1);
  const float e = emb[(size_t)tok[row] * D + d] * 22.62741699796952f;  // sqrt(512)
  const int i2  = d & ~1;
  const float div = __expf(-(float)i2 * (9.210340371976184f / (float)D)); // ln(1e4)/D
  const float ang = (float)l * div;
  const float pe  = (d & 1) ? __cosf(ang) : __sinf(ang);
  X[idx] = e + pe;
}

// Extract V from KV and transpose to (B*H*64, 1024) for contiguous B-frags.
__global__ void transpose_v(const _Float16* __restrict__ KV, _Float16* __restrict__ VT)
{
  const int idx = blockIdx.x * blockDim.x + threadIdx.x;
  if (idx >= BB * NHEAD * DKH * LSEQ) return;
  const int l  = idx & (LSEQ - 1);
  const int r  = idx >> 10;
  const int dk = r & (DKH - 1);
  const int hh = (r >> 6) & 7;
  const int b  = r >> 9;
  VT[idx] = KV[(size_t)(b * LSEQ + l) * (2 * D) + D + hh * DKH + dk];
}

__global__ void f32_to_f16(const float* __restrict__ s, _Float16* __restrict__ d, int n)
{
  const int i = blockIdx.x * blockDim.x + threadIdx.x;
  if (i < n) d[i] = (_Float16)s[i];
}

// Tiny classifier head: (4096,512) @ (10,512)^T + bc, fp32.
__global__ void classifier(const float* __restrict__ X, const float* __restrict__ Wc,
                           const float* __restrict__ bc, float* __restrict__ out)
{
  const int idx = blockIdx.x * blockDim.x + threadIdx.x;
  if (idx >= NROWS * NCLS) return;
  const int c   = idx % NCLS;
  const int row = idx / NCLS;
  const float4* xp = (const float4*)(X + (size_t)row * D);
  const float4* wp = (const float4*)(Wc + (size_t)c * D);
  float s = 0.0f;
#pragma unroll 4
  for (int i = 0; i < D / 4; ++i) {
    const float4 xv = xp[i];
    const float4 wv = wp[i];
    s += xv.x * wv.x + xv.y * wv.y + xv.z * wv.z + xv.w * wv.w;
  }
  out[idx] = s + bc[c];
}

extern "C" void kernel_launch(void* const* d_in, const int* in_sizes, int n_in,
                              void* d_out, int out_size, void* d_ws, size_t ws_size,
                              hipStream_t stream)
{
  const int*   tokens = (const int*)d_in[0];
  const float* emb  = (const float*)d_in[1];
  const float* Wq   = (const float*)d_in[2];
  const float* bq   = (const float*)d_in[3];
  const float* Wkv  = (const float*)d_in[4];
  const float* bkv  = (const float*)d_in[5];
  const float* Wo   = (const float*)d_in[6];
  const float* bo   = (const float*)d_in[7];
  const float* g1   = (const float*)d_in[8];
  const float* be1  = (const float*)d_in[9];
  const float* W1   = (const float*)d_in[10];
  const float* b1   = (const float*)d_in[11];
  const float* W2   = (const float*)d_in[12];
  const float* b2   = (const float*)d_in[13];
  const float* g2   = (const float*)d_in[14];
  const float* be2  = (const float*)d_in[15];
  const float* Wc   = (const float*)d_in[16];
  const float* bc   = (const float*)d_in[17];

  char* ws = (char*)d_ws;
  size_t off = 0;
  auto carve = [&](size_t bytes) -> void* {
    void* p = ws + off;
    off += (bytes + 255) & ~(size_t)255;
    return p;
  };
  float*    X    = (float*)   carve((size_t)NROWS * D * 4);
  _Float16* h16  = (_Float16*)carve((size_t)NROWS * D * 2);
  _Float16* q16  = (_Float16*)carve((size_t)NROWS * D * 2);
  _Float16* kv16 = (_Float16*)carve((size_t)NROWS * 2 * D * 2);
  _Float16* vt16 = (_Float16*)carve((size_t)NROWS * D * 2);
  _Float16* av16 = (_Float16*)carve((size_t)NROWS * D * 2);
  _Float16* ff16 = (_Float16*)carve((size_t)NROWS * DFF * 2);
  _Float16* wq16 = (_Float16*)carve((size_t)D * D * 2);
  _Float16* wkv16= (_Float16*)carve((size_t)2 * D * D * 2);
  _Float16* wo16 = (_Float16*)carve((size_t)D * D * 2);
  _Float16* w116 = (_Float16*)carve((size_t)DFF * D * 2);
  _Float16* w216 = (_Float16*)carve((size_t)D * DFF * 2);

  const dim3 blk(256);
  embed_pe<<<dim3((NROWS * D + 255) / 256), blk, 0, stream>>>(tokens, emb, X);

  const float inv_scale = 0.125f;  // 1/sqrt(DK=64), folded into the Q GEMM
  for (int l = 0; l < NLAYER; ++l) {
    f32_to_f16<<<(D * D + 255) / 256, blk, 0, stream>>>(Wq + (size_t)l * D * D, wq16, D * D);
    f32_to_f16<<<(2 * D * D + 255) / 256, blk, 0, stream>>>(Wkv + (size_t)l * 2 * D * D, wkv16, 2 * D * D);
    f32_to_f16<<<(D * D + 255) / 256, blk, 0, stream>>>(Wo + (size_t)l * D * D, wo16, D * D);
    f32_to_f16<<<(DFF * D + 255) / 256, blk, 0, stream>>>(W1 + (size_t)l * DFF * D, w116, DFF * D);
    f32_to_f16<<<(D * DFF + 255) / 256, blk, 0, stream>>>(W2 + (size_t)l * D * DFF, w216, D * DFF);

    ln_rows<<<NROWS, blk, 0, stream>>>(X, g1 + l * D, be1 + l * D, h16);
    // q = LN(x) @ Wq^T + bq, scaled by 1/sqrt(dk), f16 out
    gemm_wmma<0, 0, 0, 1><<<dim3(D / 128, NROWS / 128), blk, 0, stream>>>(
        h16, wq16, bq + l * D, nullptr, nullptr, q16, NROWS, D, D, inv_scale);
    // kv = LN(x) @ Wkv^T + bkv, f16 out
    gemm_wmma<0, 0, 0, 1><<<dim3(2 * D / 128, NROWS / 128), blk, 0, stream>>>(
        h16, wkv16, bkv + l * 2 * D, nullptr, nullptr, kv16, NROWS, 2 * D, D, 1.0f);
    transpose_v<<<(BB * NHEAD * DKH * LSEQ + 255) / 256, blk, 0, stream>>>(kv16, vt16);
    attn_wmma<<<dim3(LSEQ / 128, BB * NHEAD), blk, 0, stream>>>(q16, kv16, vt16, av16);
    // x = x + av @ Wo^T + bo   (fp32 residual accumulate)
    gemm_wmma<0, 1, 1, 0><<<dim3(D / 128, NROWS / 128), blk, 0, stream>>>(
        av16, wo16, bo + l * D, X, X, nullptr, NROWS, D, D, 1.0f);
    ln_rows<<<NROWS, blk, 0, stream>>>(X, g2 + l * D, be2 + l * D, h16);
    // ffn hidden = relu(LN(x) @ W1^T + b1), f16 out
    gemm_wmma<1, 0, 0, 1><<<dim3(DFF / 128, NROWS / 128), blk, 0, stream>>>(
        h16, w116, b1 + l * DFF, nullptr, nullptr, ff16, NROWS, DFF, D, 1.0f);
    // x = x + hidden @ W2^T + b2
    gemm_wmma<0, 1, 1, 0><<<dim3(D / 128, NROWS / 128), blk, 0, stream>>>(
        ff16, w216, b2 + l * D, X, X, nullptr, NROWS, D, DFF, 1.0f);
  }

  classifier<<<(NROWS * NCLS + 255) / 256, blk, 0, stream>>>(X, Wc, bc, (float*)d_out);
  (void)in_sizes; (void)n_in; (void)out_size; (void)ws_size;
}
